// MyLSTM_33904471834975
// MI455X (gfx1250) — compile-verified, matrix operand order checked by reading, not run on hardware
//
#include <hip/hip_runtime.h>
#include <hip/hip_bf16.h>

// ---------------------------------------------------------------------------
// LSTM: T=512, B=256, I=13, H=512.
// Persistent kernel, all 512 timesteps in one launch. Recurrent GEMM uses
// v_wmma_f32_16x16x32_bf16. The wave's W_hh tile (256 VGPRs of bf16
// fragments) is preloaded into registers ONCE and reused for all timesteps —
// per-step VMEM is only the h fragments. Grid barrier between steps.
// ---------------------------------------------------------------------------

#define TT 512
#define BB 256
#define II 13
#define HH 512
#define G4 (4 * HH)   // 2048
#define KITERS (HH / 32)               // 16 WMMA k-steps
#define NBLK ((BB / 32) * (HH / 32))   // 128 persistent workgroups

typedef __attribute__((ext_vector_type(16))) __bf16 v16bf;
typedef __attribute__((ext_vector_type(8)))  __bf16 v8bf;
typedef __attribute__((ext_vector_type(8)))  float  v8f;

__device__ __forceinline__ unsigned short f2bf(float f) {
    union { float f; unsigned u; } v; v.f = f;
    unsigned r = v.u + 0x7FFFu + ((v.u >> 16) & 1u);   // round-to-nearest-even
    return (unsigned short)(r >> 16);
}
__device__ __forceinline__ float bf2f(unsigned short s) {
    union { unsigned u; float f; } v; v.u = ((unsigned)s) << 16;
    return v.f;
}

// Load a 16x32 bf16 WMMA fragment for this lane.
// Per ISA layout: lane(0-15)=row, khalf=lane>>4; VGPR0-3 hold K=khalf*8..+7,
// VGPR4-7 hold K=16+khalf*8..+7 -> two contiguous 16-byte chunks 16 elems apart.
__device__ __forceinline__ v16bf load_frag(const unsigned short* p) {
    v8bf lo = *(const v8bf*)(p);
    v8bf hi = *(const v8bf*)(p + 16);
    return __builtin_shufflevector(lo, hi,
        0, 1, 2, 3, 4, 5, 6, 7, 8, 9, 10, 11, 12, 13, 14, 15);
}

// ---------------------------------------------------------------------------
// Kernel 1: convert W_hh to bf16, zero h0/c and the per-step barrier counters.
// ---------------------------------------------------------------------------
__global__ __launch_bounds__(256) void lstm_init_kernel(
    const float* __restrict__ W_hh,   // [4H, H]
    unsigned short* __restrict__ whh_bf,
    unsigned short* __restrict__ h0,
    float* __restrict__ c,
    unsigned int* __restrict__ cnt)   // [TT] barrier counters
{
    const int i = blockIdx.x * 256 + threadIdx.x;
    if (i < G4 * HH) whh_bf[i] = f2bf(W_hh[i]);
    if (i < BB * HH) { c[i] = 0.0f; h0[i] = 0; }
    if (i < TT) cnt[i] = 0u;
}

// ---------------------------------------------------------------------------
// Kernel 2: persistent LSTM. grid = 128 blocks x 128 threads (4 waves).
// Wave g (0..3) computes gate g's 32x32 pre-activation tile each step with
// WMMA; its W_hh fragments live in VGPRs for the whole kernel.
// ---------------------------------------------------------------------------
__global__ __launch_bounds__(128, 1) void lstm_persistent_kernel(
    const float* __restrict__ x,          // [T,B,I]
    const float* __restrict__ W_ih,       // [4H,I]
    const float* __restrict__ b_ih,       // [4H]
    const float* __restrict__ b_hh,       // [4H]
    const unsigned short* __restrict__ whh,   // [4H,H] bf16
    unsigned short* __restrict__ h0,          // [B,H] bf16 (even-step input)
    unsigned short* __restrict__ h1,          // [B,H] bf16 (odd-step input)
    float* __restrict__ c,                    // [B,H] f32
    unsigned short* __restrict__ hs,          // [T,B,H] bf16 history
    unsigned int* __restrict__ cnt)           // [TT] barrier counters
{
    __shared__ float xs[32][14];        // x tile [32 rows][13] (padded)
    __shared__ float gs[4][32][33];     // activated gate tiles (padded)

    const int tid  = threadIdx.x;
    const int gate = tid >> 5;          // wave id == gate index (i,f,g,o)
    const int lane = tid & 31;
    const int lq   = lane & 15;
    const int hl   = lane >> 4;
    const int bm   = blockIdx.x & 7;    // 8 tiles over B
    const int hn   = blockIdx.x >> 3;   // 16 tiles over H

    // ---- loop-invariant: W_ih rows + biases for this lane's two N columns ----
    float wv[2][II];
    float bias[2];
#pragma unroll
    for (int ns = 0; ns < 2; ++ns) {
        const int gc = gate * HH + hn * 32 + ns * 16 + lq;
#pragma unroll
        for (int i = 0; i < II; ++i) wv[ns][i] = W_ih[gc * II + i];
        bias[ns] = b_ih[gc] + b_hh[gc];
    }

    // ---- preload this wave's W_hh fragments into VGPRs (timestep-invariant;
    //      2 n-tiles x 16 k-steps x v16bf = 256 VGPRs, held for all 512 steps)
    const unsigned short* bB0 = whh + (size_t)(gate * HH + hn * 32 + lq) * HH + hl * 8;
    const unsigned short* bB1 = whh + (size_t)(gate * HH + hn * 32 + 16 + lq) * HH + hl * 8;
    v16bf wf0[KITERS], wf1[KITERS];
#pragma unroll
    for (int kk = 0; kk < KITERS; ++kk) {
        wf0[kk] = load_frag(bB0 + kk * 32);
        wf1[kk] = load_frag(bB1 + kk * 32);
    }

    // ---- loop-invariant A-fragment offsets ----
    const int aOff0 = (bm * 32 + lq) * HH + hl * 8;        // A tile rows 0..15
    const int aOff1 = (bm * 32 + 16 + lq) * HH + hl * 8;   // A tile rows 16..31

    for (int t = 0; t < TT; ++t) {
        const unsigned short* hprev = (t & 1) ? h1 : h0;
        unsigned short*       hnext = (t & 1) ? h0 : h1;

        // ---- stage x[t, bm*32 .. +31, :] into LDS ----
        for (int idx = tid; idx < 32 * II; idx += 128) {
            int r = idx / II, ci = idx % II;
            xs[r][ci] = x[(size_t)t * BB * II + (size_t)(bm * 32 + r) * II + ci];
        }
        __syncthreads();

        // ---- accumulator init: bias + x @ W_ih^T (K=13 scalar FMAs) ----
        // C-layout: VGPR r <-> M = r + 8*hl (+16*ms), N = lq (+16*ns)
        v8f acc[2][2];
#pragma unroll
        for (int ns = 0; ns < 2; ++ns) {
#pragma unroll
            for (int ms = 0; ms < 2; ++ms) {
#pragma unroll
                for (int r = 0; r < 8; ++r) {
                    const int m = ms * 16 + r + hl * 8;
                    float s = bias[ns];
#pragma unroll
                    for (int i = 0; i < II; ++i) s += xs[m][i] * wv[ns][i];
                    acc[ms][ns][r] = s;
                }
            }
        }

        // ---- main K loop: h_prev @ W_hh^T; weights already in VGPRs ----
        const unsigned short* aB0 = hprev + aOff0;
        const unsigned short* aB1 = hprev + aOff1;
#pragma unroll
        for (int kk = 0; kk < KITERS; ++kk) {
            v16bf a0 = load_frag(aB0 + kk * 32);
            v16bf a1 = load_frag(aB1 + kk * 32);
            acc[0][0] = __builtin_amdgcn_wmma_f32_16x16x32_bf16(
                false, a0, false, wf0[kk], (short)0, acc[0][0], false, false);
            acc[0][1] = __builtin_amdgcn_wmma_f32_16x16x32_bf16(
                false, a0, false, wf1[kk], (short)0, acc[0][1], false, false);
            acc[1][0] = __builtin_amdgcn_wmma_f32_16x16x32_bf16(
                false, a1, false, wf0[kk], (short)0, acc[1][0], false, false);
            acc[1][1] = __builtin_amdgcn_wmma_f32_16x16x32_bf16(
                false, a1, false, wf1[kk], (short)0, acc[1][1], false, false);
        }

        // ---- activation (uniform per wave) + exchange through LDS ----
#pragma unroll
        for (int ms = 0; ms < 2; ++ms) {
#pragma unroll
            for (int ns = 0; ns < 2; ++ns) {
#pragma unroll
                for (int r = 0; r < 8; ++r) {
                    const int m = ms * 16 + r + hl * 8;
                    const int n = ns * 16 + lq;
                    float v = acc[ms][ns][r];
                    v = (gate == 2) ? tanhf(v) : 1.0f / (1.0f + __expf(-v));
                    gs[gate][m][n] = v;
                }
            }
        }
        __syncthreads();

        // ---- elementwise cell/hidden update (1024 elems / 128 threads) ----
        // Ownership of c[idx] is step-invariant -> no cross-block sync needed.
        for (int e = tid; e < 32 * 32; e += 128) {
            const int m = e >> 5, n = e & 31;
            const size_t idx = (size_t)(bm * 32 + m) * HH + (hn * 32 + n);
            const float iv = gs[0][m][n];
            const float fv = gs[1][m][n];
            const float gv = gs[2][m][n];
            const float ov = gs[3][m][n];
            const float cn = fv * c[idx] + iv * gv;
            const float hv = ov * tanhf(cn);
            c[idx] = cn;
            const unsigned short hb = f2bf(hv);
            hnext[idx] = hb;
            hs[(size_t)t * BB * HH + idx] = hb;
        }

        // ---- device-scope grid barrier before next timestep ----
        __threadfence();
        if (tid == 0) {
            __hip_atomic_fetch_add(&cnt[t], 1u, __ATOMIC_RELEASE,
                                   __HIP_MEMORY_SCOPE_AGENT);
            while (__hip_atomic_load(&cnt[t], __ATOMIC_ACQUIRE,
                                     __HIP_MEMORY_SCOPE_AGENT) < (unsigned)NBLK)
                __builtin_amdgcn_s_sleep(1);
        }
        __syncthreads();
    }
}

// ---------------------------------------------------------------------------
// Kernel 3: y[t,b] = relu(hs[t,b,:]) . W_out + b_out   (one wave per output)
// ---------------------------------------------------------------------------
__global__ __launch_bounds__(128) void lstm_out_kernel(
    const unsigned short* __restrict__ hs,  // [T*B, H] bf16
    const float* __restrict__ W_out,        // [H]
    const float* __restrict__ b_out,        // [1]
    float* __restrict__ y,                  // [T*B]
    int total)
{
    const int wid  = blockIdx.x * 4 + (threadIdx.x >> 5);
    const int lane = threadIdx.x & 31;
    if (wid >= total) return;
    const unsigned short* row = hs + (size_t)wid * HH;
    float s = 0.0f;
    for (int j = lane; j < HH; j += 32) {
        float h = bf2f(row[j]);
        h = h > 0.0f ? h : 0.0f;
        s += h * W_out[j];
    }
#pragma unroll
    for (int off = 16; off > 0; off >>= 1)
        s += __shfl_down(s, off, 32);
    if (lane == 0) y[wid] = s + b_out[0];
}

// ---------------------------------------------------------------------------
extern "C" void kernel_launch(void* const* d_in, const int* in_sizes, int n_in,
                              void* d_out, int out_size, void* d_ws, size_t ws_size,
                              hipStream_t stream) {
    (void)in_sizes; (void)n_in; (void)out_size; (void)ws_size;

    const float* x     = (const float*)d_in[0];
    const float* W_ih  = (const float*)d_in[1];
    const float* W_hh  = (const float*)d_in[2];
    const float* b_ih  = (const float*)d_in[3];
    const float* b_hh  = (const float*)d_in[4];
    const float* W_out = (const float*)d_in[5];
    const float* b_out = (const float*)d_in[6];
    float* y = (float*)d_out;

    // workspace layout (bytes)
    char* ws = (char*)d_ws;
    unsigned short* whh_bf = (unsigned short*)(ws);                    // 2 MiB
    unsigned short* h0     = (unsigned short*)(ws + 2097152);          // 256 KiB
    unsigned short* h1     = (unsigned short*)(ws + 2359296);          // 256 KiB
    float*          c      = (float*)(ws + 2621440);                   // 512 KiB
    unsigned short* hs     = (unsigned short*)(ws + 3145728);          // 128 MiB
    unsigned int*   cnt    = (unsigned int*)(ws + 3145728 + (size_t)TT * BB * HH * 2);

    // 1) convert W_hh to bf16, zero h0/c, reset barrier counters
    lstm_init_kernel<<<(G4 * HH + 255) / 256, 256, 0, stream>>>(
        W_hh, whh_bf, h0, c, cnt);

    // 2) one persistent kernel runs all 512 timesteps
    lstm_persistent_kernel<<<NBLK, 128, 0, stream>>>(
        x, W_ih, b_ih, b_hh, whh_bf, h0, h1, c, hs, cnt);

    // 3) output projection
    lstm_out_kernel<<<(TT * BB) / 4, 128, 0, stream>>>(hs, W_out, b_out, y, TT * BB);
}
